// FullFourierBlockN_17549236371538
// MI455X (gfx1250) — matrix operand order repaired and checked by reading, not compiled, not run on hardware
//
#include <hip/hip_runtime.h>
#include <stdint.h>

// ---------------------------------------------------------------------------
// FNO spectral block on MI455X (gfx1250): everything as bf16 WMMA GEMMs.
//   B=32, H=8, CI=CO=64, S0=32, S1=64, F1=33 (padded to 64), M=256.
// Memory-bound (~536MB -> ~23us @ 23.3TB/s); GEMM FLOPs (~30G) trivial on WMMA.
// This revision: TDM async tensor load for the x slices, dual-accumulator
// WMMA interleaving (fills hazard slots), LDS tile-transpose weight prep
// (the 67MB f32 weight stream is read once, coalesced on both sides).
// ---------------------------------------------------------------------------

#define Bn   32
#define Hn   8
#define CIn  64
#define COn  64
#define S0n  32
#define S1n  64
#define F1n  33
#define Mn   256
#define CELL (S0n * S1n)   // 2048 per slice

typedef __attribute__((ext_vector_type(16))) __bf16       v16bf;
typedef __attribute__((ext_vector_type(8)))  float        v8f;
typedef __attribute__((ext_vector_type(4)))  unsigned int v4u;
typedef __attribute__((ext_vector_type(8)))  int          v8i;
typedef __attribute__((ext_vector_type(4)))  int          v4i;

// ---------------- WMMA fragment helpers (wave32, CDNA5 layouts) ------------

static __device__ inline v8f zero8() {
  v8f z;
#pragma unroll
  for (int i = 0; i < 8; ++i) z[i] = 0.0f;
  return z;
}

// A-matrix 16x32 bf16, row-major source with leading-dim lda.
// lane<16: M=lane, K={0..7,16..23}; lane>=16: M=lane-16, K={8..15,24..31}.
static __device__ inline v16bf load_a(const __bf16* A, int lda, int row0,
                                      int k0, int lane) {
  const int m  = lane & 15;
  const int kh = (lane >> 4) * 8;
  v16bf a;
#pragma unroll
  for (int j = 0; j < 16; ++j) {
    const int k = ((j < 8) ? j : j + 8) + kh;
    a[j] = A[(row0 + m) * lda + k0 + k];
  }
  return a;
}

// Same, but converting from an f32 source (e.g. TDM-staged x slice).
static __device__ inline v16bf load_a_f32(const float* A, int lda, int row0,
                                          int k0, int lane) {
  const int m  = lane & 15;
  const int kh = (lane >> 4) * 8;
  v16bf a;
#pragma unroll
  for (int j = 0; j < 16; ++j) {
    const int k = ((j < 8) ? j : j + 8) + kh;
    a[j] = (__bf16)A[(row0 + m) * lda + k0 + k];
  }
  return a;
}

// B-matrix 32x16 bf16; source stored TRANSPOSED: Bt[n][k], row length ldb.
// lane<16: N=lane, K={0..7,16..23}; lane>=16: N=lane-16, K={8..15,24..31}.
static __device__ inline v16bf load_b(const __bf16* Bt, int ldb, int n0,
                                      int k0, int lane) {
  const int n  = lane & 15;
  const int kb = (lane >> 4) * 8;
  const __bf16* p = Bt + (n0 + n) * ldb + k0 + kb;
  v16bf b;
#pragma unroll
  for (int j = 0; j < 16; ++j) b[j] = p[(j < 8) ? j : j + 8];
  return b;
}

// bf16 WMMA A/B-negation is not supported by the ISA (NEG is C-only) -> VALU.
static __device__ inline v16bf neg_frag(v16bf x) {
  v16bf r;
#pragma unroll
  for (int j = 0; j < 16; ++j) r[j] = (__bf16)(-(float)x[j]);
  return r;
}

static __device__ inline v8f wmma_bf16(v16bf a, v16bf b, v8f c) {
  // emits v_wmma_f32_16x16x32_bf16
  return __builtin_amdgcn_wmma_f32_16x16x32_bf16(false, a, false, b,
                                                 (short)0, c, false, false);
}

// C/D: VGPR v holds rows M=v (lanes 0-15) and M=v+8 (lanes 16-31), N=lane%16.
static __device__ inline void store_d_bf16(__bf16* D, int ldd, int row0,
                                           int col0, int lane, v8f acc) {
  const int c  = lane & 15;
  const int kh = (lane >> 4) * 8;
#pragma unroll
  for (int v = 0; v < 8; ++v)
    D[(row0 + v + kh) * ldd + col0 + c] = (__bf16)acc[v];
}

// Store transposed: element (m,n) -> Dt[n*ldt + m]
static __device__ inline void store_d_bf16_T(__bf16* Dt, int ldt, int row0,
                                             int col0, int lane, v8f acc) {
  const int c  = lane & 15;
  const int kh = (lane >> 4) * 8;
#pragma unroll
  for (int v = 0; v < 8; ++v)
    Dt[(col0 + c) * ldt + row0 + v + kh] = (__bf16)acc[v];
}

static __device__ inline void store_d_f32(float* D, int ldd, int row0,
                                          int col0, int lane, v8f acc) {
  const int c  = lane & 15;
  const int kh = (lane >> 4) * 8;
#pragma unroll
  for (int v = 0; v < 8; ++v)
    D[(row0 + v + kh) * ldd + col0 + c] = acc[v];
}

// ---------------- K0a: twiddle matrices + inverse-mask init ----------------
// A1t[j1][n1] = e^{-2pi i j1 n1/64}/8 (rows j1>=33 zero)     (forward, S1 dim)
// A0 [j0][n0] = e^{-2pi i j0 n0/32}/sqrt(32)                 (forward, S0 dim)
// B0 [n0][j0] = e^{+2pi i n0 j0/32}/sqrt(32)                 (inverse, S0 dim)
// B1t[n1][j1] = c_j1 * cos(2pi j1 n1/64)/8, c={1,2,...,2,1}  (real irfft, S1)
__global__ void fno_init_twiddles(__bf16* A1tr, __bf16* A1ti, __bf16* A0r,
                                  __bf16* A0i, __bf16* B0r, __bf16* B0i,
                                  __bf16* B1ct, int* invmask) {
  const int t = blockIdx.x * blockDim.x + threadIdx.x;
  const float W64 = 6.283185307179586f / 64.0f;
  const float W32 = 6.283185307179586f / 32.0f;
  if (t < 64 * 64) {
    { // A1t (row j1, col n1)
      const int j1 = t >> 6, n1 = t & 63;
      float re = 0.f, im = 0.f;
      if (j1 <= 32) {
        const int p = (j1 * n1) & 63;
        re =  cosf(p * W64) * 0.125f;
        im = -sinf(p * W64) * 0.125f;
      }
      A1tr[t] = (__bf16)re;
      A1ti[t] = (__bf16)im;
    }
    { // B1t (row n1, col j1)
      const int n1 = t >> 6, j1 = t & 63;
      float v = 0.f;
      if (j1 <= 32) {
        const float c = (j1 == 0 || j1 == 32) ? 1.0f : 2.0f;
        const int p = (j1 * n1) & 63;
        v = c * cosf(p * W64) * 0.125f;
      }
      B1ct[t] = (__bf16)v;
    }
  }
  if (t < 32 * 32) {
    const int r = t >> 5, k = t & 31;
    const int p = (r * k) & 31;
    const float c = cosf(p * W32), s = sinf(p * W32);
    const float n32 = 0.17677669529663687f; // 1/sqrt(32)
    A0r[t] = (__bf16)( c * n32);
    A0i[t] = (__bf16)(-s * n32);
    B0r[t] = (__bf16)( c * n32);
    B0i[t] = (__bf16)( s * n32);
  }
  if (t < S0n * F1n) invmask[t] = -1;
}

__global__ void fno_init_invmask(const int* __restrict__ mask, int* invmask) {
  const int t = threadIdx.x;
  if (t < Mn) invmask[mask[t]] = t;
}

// ---------------- K2: weight prep, LDS tile transpose ----------------------
// (H,CI,CO,M) f32 -> [h][l][o][i] bf16.  The 67MB weight stream is read
// once (wr+wi, f32), so both sides must be coalesced: 32x32 tile
// over (i,l) per block, transposed through padded LDS.
__global__ void __launch_bounds__(256)
fno_weight_prep(const float* __restrict__ wr, const float* __restrict__ wi,
                __bf16* __restrict__ Wtr, __bf16* __restrict__ Wti) {
  __shared__ float tile[32][33];
  const int tid = threadIdx.x;
  const int r = tid >> 5, c = tid & 31;
  const int lt = blockIdx.x & 7;          // M/32 tiles
  const int it = (blockIdx.x >> 3) & 1;   // CI/32 tiles
  const int o  = (blockIdx.x >> 4) & 63;
  const int h  = blockIdx.x >> 10;
  const float* srcs[2] = {wr, wi};
  __bf16*      dsts[2] = {Wtr, Wti};
#pragma unroll
  for (int p = 0; p < 2; ++p) {
    const float* s = srcs[p];
    __bf16* d = dsts[p];
    for (int rr = r; rr < 32; rr += 8)  // rr = local i ; c = local l (coalesced)
      tile[rr][c] =
          s[(((size_t)h * CIn + it * 32 + rr) * COn + o) * Mn + lt * 32 + c];
    __syncthreads();
    for (int rr = r; rr < 32; rr += 8)  // rr = local l ; c = local i (coalesced)
      d[(((size_t)h * Mn + lt * 32 + rr) * COn + o) * CIn + it * 32 + c] =
          (__bf16)tile[c][rr];
    __syncthreads();
  }
}

// ---------------- K1: fused forward rfft2-as-GEMM + mode gather ------------
// One WG per (b,h,ci) slice; 8 waves.  T = x*A1 ; X = A0*T ; gather 256 modes.
// x slice is staged to LDS by the Tensor Data Mover (async, TENSORcnt).
__global__ void __launch_bounds__(256)
fno_forward(const float* __restrict__ x, const int* __restrict__ mask,
            const __bf16* __restrict__ A1tr, const __bf16* __restrict__ A1ti,
            const __bf16* __restrict__ A0r, const __bf16* __restrict__ A0i,
            __bf16* __restrict__ xmr, __bf16* __restrict__ xmi) {
  __shared__ float  xlf[S0n * S1n];  // 32x64 f32 slice (TDM destination)
  __shared__ __bf16 Ttr[S1n * S0n];  // T transposed [j1][n0]
  __shared__ __bf16 Tti[S1n * S0n];
  __shared__ __bf16 Xr[S0n * S1n];   // X row-major [j0][j1]
  __shared__ __bf16 Xi[S0n * S1n];

  const int tid  = threadIdx.x;
  const int wave = tid >> 5;
  const int lane = tid & 31;
  const int ci = blockIdx.x % CIn;
  const int h  = (blockIdx.x / CIn) % Hn;
  const int b  = blockIdx.x / (CIn * Hn);

  const float* xg = x + (size_t)blockIdx.x * CELL;

#if __has_builtin(__builtin_amdgcn_tensor_load_to_lds) && \
    __has_builtin(__builtin_amdgcn_s_wait_tensorcnt)
  if (wave == 0) {
    // D# group0: count=1 | lds_addr | global_addr[56:0] | type=2 ("image")
    const uint64_t ga = (uint64_t)(uintptr_t)xg;
    v4u g0;
    g0[0] = 1u;
    g0[1] = (unsigned)(uintptr_t)&xlf[0];            // LDS byte offset
    g0[2] = (unsigned)ga;
    g0[3] = (unsigned)((ga >> 32) & 0x01FFFFFFu) | (2u << 30);
    // D# group1: data_size=4B | tensor 64x32 | tile 64x32 | dim0_stride=64
    v8i g1;
    g1[0] = (int)(2u << 16);   // data_size=2 (4 bytes), no multicast/pad/iter
    g1[1] = (int)(64u << 16);  // tensor_dim0 = 64
    g1[2] = (int)(32u << 16);  // tensor_dim1 = 32
    g1[3] = (int)(64u << 16);  // tile_dim0 = 64
    g1[4] = 32;                // tile_dim1 = 32 (tile_dim2 = 0)
    g1[5] = 64;                // tensor_dim0_stride = 64
    g1[6] = 0;
    g1[7] = 0;
    v4i z4 = {0, 0, 0, 0};
#if __clang_major__ >= 23
    v8i z8 = {0, 0, 0, 0, 0, 0, 0, 0};
    __builtin_amdgcn_tensor_load_to_lds(g0, g1, z4, z4, z8, 0);
#else
    __builtin_amdgcn_tensor_load_to_lds(g0, g1, z4, z4, 0);
#endif
    __builtin_amdgcn_s_wait_tensorcnt(0);  // s_wait_tensorcnt 0
  }
#else
  for (int i = tid; i < CELL; i += 256) xlf[i] = xg[i];
#endif
  __syncthreads();

  // GEMM1: T(32x64,cplx) = x(32x64) * A1(64x64,cplx).
  // Each wave: one (part,nt) pair, BOTH m-tiles -> 2 independent WMMA chains
  // sharing the B fragment (fills the WMMA->WMMA hazard slots).
  {
    const int part = wave & 1;
    const int nt   = (wave >> 1) & 3;
    const __bf16* Bt = part ? A1ti : A1tr;
    v8f acc0 = zero8(), acc1 = zero8();
#pragma unroll
    for (int k0 = 0; k0 < 64; k0 += 32) {
      v16bf bb = load_b(Bt, 64, nt * 16, k0, lane);
      v16bf a0 = load_a_f32(xlf, 64, 0, k0, lane);
      v16bf a1 = load_a_f32(xlf, 64, 16, k0, lane);
      acc0 = wmma_bf16(a0, bb, acc0);
      acc1 = wmma_bf16(a1, bb, acc1);
    }
    __bf16* Td = part ? Tti : Ttr;
    store_d_bf16_T(Td, 32, 0, nt * 16, lane, acc0);
    store_d_bf16_T(Td, 32, 16, nt * 16, lane, acc1);
  }
  __syncthreads();

  // GEMM2: X(32x64,cplx) = A0(32x32,cplx) * T; K=32, dual m-tile interleave.
  {
    const int part = wave & 1;
    const int nt   = (wave >> 1) & 3;
    if (part == 0) { // Xr = A0r*Tr - A0i*Ti
      v16bf Br  = load_b(Ttr, 32, nt * 16, 0, lane);
      v16bf Bi  = load_b(Tti, 32, nt * 16, 0, lane);
      v16bf ar0 = load_a(A0r, 32, 0, 0, lane);
      v16bf ar1 = load_a(A0r, 32, 16, 0, lane);
      v16bf ai0 = neg_frag(load_a(A0i, 32, 0, 0, lane));
      v16bf ai1 = neg_frag(load_a(A0i, 32, 16, 0, lane));
      v8f acc0 = wmma_bf16(ar0, Br, zero8());
      v8f acc1 = wmma_bf16(ar1, Br, zero8());
      acc0 = wmma_bf16(ai0, Bi, acc0);
      acc1 = wmma_bf16(ai1, Bi, acc1);
      store_d_bf16(Xr, 64, 0, nt * 16, lane, acc0);
      store_d_bf16(Xr, 64, 16, nt * 16, lane, acc1);
    } else {         // Xi = A0r*Ti + A0i*Tr
      v16bf Br  = load_b(Ttr, 32, nt * 16, 0, lane);
      v16bf Bi  = load_b(Tti, 32, nt * 16, 0, lane);
      v16bf ar0 = load_a(A0r, 32, 0, 0, lane);
      v16bf ar1 = load_a(A0r, 32, 16, 0, lane);
      v16bf ai0 = load_a(A0i, 32, 0, 0, lane);
      v16bf ai1 = load_a(A0i, 32, 16, 0, lane);
      v8f acc0 = wmma_bf16(ar0, Bi, zero8());
      v8f acc1 = wmma_bf16(ar1, Bi, zero8());
      acc0 = wmma_bf16(ai0, Br, acc0);
      acc1 = wmma_bf16(ai1, Br, acc1);
      store_d_bf16(Xi, 64, 0, nt * 16, lane, acc0);
      store_d_bf16(Xi, 64, 16, nt * 16, lane, acc1);
    }
  }
  __syncthreads();

  // Gather the 256 active modes: xm[h][l][b][ci]  (xm stays L2-resident).
  {
    const int l   = tid;           // 256 threads == 256 modes
    const int idx = mask[l];
    const int j0  = idx / F1n;
    const int j1  = idx - j0 * F1n;
    const size_t dst = (((size_t)h * Mn + l) * Bn + b) * CIn + ci;
    xmr[dst] = Xr[j0 * 64 + j1];
    xmi[dst] = Xi[j0 * 64 + j1];
  }
}

// ---------------- K3: per-(h,l) complex channel GEMM  om = xm * W ----------
// One wave per 16x16 output tile; 16384 waves total; 8 WMMAs each (K=64),
// alternating between the R and I accumulators (independent chains).
__global__ void __launch_bounds__(256)
fno_mode_gemm(const __bf16* __restrict__ xmr, const __bf16* __restrict__ xmi,
              const __bf16* __restrict__ Wtr, const __bf16* __restrict__ Wti,
              __bf16* __restrict__ omr, __bf16* __restrict__ omi) {
  const int lane = threadIdx.x & 31;
  const int gw   = blockIdx.x * 8 + (threadIdx.x >> 5);
  const int nt = gw & 3;
  const int mt = (gw >> 2) & 1;
  const int l  = (gw >> 3) & 255;
  const int h  = gw >> 11;

  const size_t hl = (size_t)h * Mn + l;
  const __bf16* ar = xmr + hl * (Bn * CIn);   // 32x64, lda=64
  const __bf16* ai = xmi + hl * (Bn * CIn);
  const __bf16* br = Wtr + hl * (COn * CIn);  // [o][i], ldb=64
  const __bf16* bi = Wti + hl * (COn * CIn);
  __builtin_prefetch(br + COn * CIn, 0, 0);   // global_prefetch next mode's W

  v8f accR = zero8(), accI = zero8();
#pragma unroll
  for (int k0 = 0; k0 < 64; k0 += 32) {
    v16bf Ar  = load_a(ar, 64, mt * 16, k0, lane);
    v16bf Ai  = load_a(ai, 64, mt * 16, k0, lane);
    v16bf Br  = load_b(br, 64, nt * 16, k0, lane);
    v16bf Bi  = load_b(bi, 64, nt * 16, k0, lane);
    v16bf Ain = neg_frag(Ai);
    accR = wmma_bf16(Ar, Br, accR);
    accI = wmma_bf16(Ar, Bi, accI);
    accR = wmma_bf16(Ain, Bi, accR);
    accI = wmma_bf16(Ai, Br, accI);
  }
  store_d_bf16(omr + hl * (Bn * COn), 64, mt * 16, nt * 16, lane, accR);
  store_d_bf16(omi + hl * (Bn * COn), 64, mt * 16, nt * 16, lane, accI);
}

// ---------------- K5: fused inverse ---------------------------------------
// Hermitian-symmetrize + ifft(j0) collapses to Re(B0*O):
//   V = B0r*Or - B0i*Oi  (real),  then  y = V * B1^T (weighted cosines).
__global__ void __launch_bounds__(256)
fno_inverse(const __bf16* __restrict__ omr, const __bf16* __restrict__ omi,
            const int* __restrict__ invmask,
            const __bf16* __restrict__ B0r, const __bf16* __restrict__ B0i,
            const __bf16* __restrict__ B1ct, float* __restrict__ out) {
  __shared__ __bf16 Otr[S1n * S0n];  // O transposed [j1][j0]
  __shared__ __bf16 Oti[S1n * S0n];
  __shared__ __bf16 Vl[S0n * S1n];   // V row-major [n0][j1]

  const int tid  = threadIdx.x;
  const int wave = tid >> 5;
  const int lane = tid & 31;
  const int co = blockIdx.x % COn;
  const int h  = (blockIdx.x / COn) % Hn;
  const int b  = blockIdx.x / (COn * Hn);

  // Scatter sparse spectrum into LDS (zeros elsewhere); om is L2-resident.
  for (int c = tid; c < CELL; c += 256) {
    const int j0 = c >> 6, j1 = c & 63;
    float vr = 0.f, vi = 0.f;
    if (j1 < F1n) {
      const int inv = invmask[j0 * F1n + j1];
      if (inv >= 0) {
        const size_t s = (((size_t)h * Mn + inv) * Bn + b) * COn + co;
        vr = (float)omr[s];
        vi = (float)omi[s];
      }
    }
    Otr[j1 * 32 + j0] = (__bf16)vr;
    Oti[j1 * 32 + j0] = (__bf16)vi;
  }
  __syncthreads();

  // V(32x64) = B0r*Or - B0i*Oi ; 8 tiles / 8 waves, K=32.
  {
    const int mt = wave >> 2, nt = wave & 3;
    v8f acc = zero8();
    acc = wmma_bf16(load_a(B0r, 32, mt * 16, 0, lane),
                    load_b(Otr, 32, nt * 16, 0, lane), acc);
    acc = wmma_bf16(neg_frag(load_a(B0i, 32, mt * 16, 0, lane)),
                    load_b(Oti, 32, nt * 16, 0, lane), acc);
    store_d_bf16(Vl, 64, mt * 16, nt * 16, lane, acc);
  }
  __syncthreads();

  // y(32x64) = V * B1^T ; K=64 (cols 33..63 of V are zero by construction).
  {
    const int mt = wave >> 2, nt = wave & 3;
    v8f acc = zero8();
#pragma unroll
    for (int k0 = 0; k0 < 64; k0 += 32) {
      v16bf a = load_a(Vl, 64, mt * 16, k0, lane);
      v16bf bb = load_b(B1ct, 64, nt * 16, k0, lane);
      acc = wmma_bf16(a, bb, acc);
    }
    store_d_f32(out + (size_t)blockIdx.x * CELL, 64, mt * 16, nt * 16, lane,
                acc);
  }
}

// ---------------------------------------------------------------------------

extern "C" void kernel_launch(void* const* d_in, const int* in_sizes, int n_in,
                              void* d_out, int out_size, void* d_ws,
                              size_t ws_size, hipStream_t stream) {
  (void)in_sizes; (void)n_in; (void)out_size; (void)ws_size;
  const float* x    = (const float*)d_in[0];
  const float* wre  = (const float*)d_in[1];
  const float* wim  = (const float*)d_in[2];
  const int*   mask = (const int*)d_in[3];
  float* out = (float*)d_out;

  char* w = (char*)d_ws;
  // twiddles + LUT (bf16 / int32)
  __bf16* A1tr = (__bf16*)(w + 0);
  __bf16* A1ti = (__bf16*)(w + 8192);
  __bf16* A0r  = (__bf16*)(w + 16384);
  __bf16* A0i  = (__bf16*)(w + 18432);
  __bf16* B0r  = (__bf16*)(w + 20480);
  __bf16* B0i  = (__bf16*)(w + 22528);
  __bf16* B1ct = (__bf16*)(w + 24576);
  int*    inv  = (int*)(w + 32768);
  // large intermediates (bf16), each 8 MB
  const size_t SEG = 8388608;
  __bf16* xmr = (__bf16*)(w + 40960);
  __bf16* xmi = (__bf16*)(w + 40960 + 1 * SEG);
  __bf16* Wtr = (__bf16*)(w + 40960 + 2 * SEG);
  __bf16* Wti = (__bf16*)(w + 40960 + 3 * SEG);
  __bf16* omr = (__bf16*)(w + 40960 + 4 * SEG);
  __bf16* omi = (__bf16*)(w + 40960 + 5 * SEG);

  fno_init_twiddles<<<16, 256, 0, stream>>>(A1tr, A1ti, A0r, A0i, B0r, B0i,
                                            B1ct, inv);
  fno_init_invmask<<<1, 256, 0, stream>>>(mask, inv);
  fno_weight_prep<<<Hn * COn * 2 * 8, 256, 0, stream>>>(wre, wim, Wtr, Wti);
  fno_forward<<<Bn * Hn * CIn, 256, 0, stream>>>(x, mask, A1tr, A1ti, A0r,
                                                 A0i, xmr, xmi);
  fno_mode_gemm<<<(Hn * Mn * 8) / 8, 256, 0, stream>>>(xmr, xmi, Wtr, Wti,
                                                       omr, omi);
  fno_inverse<<<Bn * Hn * COn, 256, 0, stream>>>(omr, omi, inv, B0r, B0i,
                                                 B1ct, out);
}
